// FFN_36867999269234
// MI455X (gfx1250) — matrix operand order repaired and checked by reading, not compiled, not run on hardware
//
#include <hip/hip_runtime.h>
#include <stdint.h>

typedef __attribute__((ext_vector_type(16))) _Float16 v16h;
typedef __attribute__((ext_vector_type(8)))  float    v8f;
typedef __attribute__((ext_vector_type(2)))  __fp16   fp16x2;
typedef __attribute__((ext_vector_type(2)))  unsigned int u2t;

union HV {
    v16h     h;
    uint32_t u[8];
};

__device__ __forceinline__ uint32_t pk2h(float a, float b) {
    union { _Float16 h[2]; uint32_t u; } c;
    c.h[0] = (_Float16)a;
    c.h[1] = (_Float16)b;
    return c.u;
}

// single-instruction f32x2 -> packed f16 (v_cvt_pk_rtz_f16_f32)
__device__ __forceinline__ uint32_t pkrtz(float a, float b) {
    union { fp16x2 h; uint32_t u; } c;
    c.h = __builtin_amdgcn_cvt_pkrtz(a, b);
    return c.u;
}

// packed f16 ReLU: one v_pk_max_num_f16 against +0
__device__ __forceinline__ uint32_t relu_pk16(uint32_t h2) {
    uint32_t r;
    asm("v_pk_max_num_f16 %0, %1, 0" : "=v"(r) : "v"(h2));
    return r;
}

__device__ __forceinline__ uint32_t swapx16_u(uint32_t v) {
    return (uint32_t)__builtin_amdgcn_ds_swizzle((int)v, 0x401f);
}

// Half-wave merge: on return
//   a = { lanes 0-15: old a (own),        lanes 16-31: old b from lane-16 }
//   b = { lanes 0-15: old a from lane+16, lanes 16-31: old b (own)        }
__device__ __forceinline__ void plswap(uint32_t& a, uint32_t& b) {
#if __has_builtin(__builtin_amdgcn_permlane16_swap)
    u2t r = __builtin_amdgcn_permlane16_swap(a, b, false, false);
    a = r.x;
    b = r.y;
#else
    const bool isLo = ((threadIdx.x & 31) >> 4) == 0;
    uint32_t sa = swapx16_u(a);
    uint32_t sb = swapx16_u(b);
    uint32_t na = isLo ? a : sb;
    uint32_t nb = isLo ? sa : b;
    a = na;
    b = nb;
#endif
}

__device__ __forceinline__ uint32_t f2u(float f) {
    union { float f; uint32_t u; } c; c.f = f; return c.u;
}
__device__ __forceinline__ float u2f(uint32_t u) {
    union { float f; uint32_t u; } c; c.u = u; return c.f;
}

__global__ __launch_bounds__(256) void ffn_ln_mlp_kernel(
    const float* __restrict__ x,
    const float* __restrict__ ln_g,
    const float* __restrict__ ln_b,
    const float* __restrict__ w1,   // (64,16) row-major
    const float* __restrict__ b1,   // (64)
    const float* __restrict__ w2,   // (16,64) row-major
    const float* __restrict__ b2,   // (16)
    float* __restrict__ out,
    int n_tokens)
{
    const int lane = threadIdx.x & 31;
    const int wv   = threadIdx.x >> 5;
    const int lo16 = lane & 15;
    const int hi   = lane >> 4;          // 0 or 1

    // ---- constant operands, register resident for the whole wave ----
    // fc1: D[d][t] = sum_h w1[d][h]*y[t][h];  A1_b[m=d_loc][k=h] = w1[16b+m][h]
    // A f16 layout: lane la: elems 0..15 -> K = 16*(la/16)+0..15 (zero pad:
    // H=16 < K=32, lanes 16..31 zero). Zero A rows K>=16 means the
    // B operand's lanes 16..31 (K=16..31) may hold anything.
    HV a1[4];
    #pragma unroll
    for (int b = 0; b < 4; ++b) {
        const float* wr = w1 + (b * 16 + lo16) * 16;
        #pragma unroll
        for (int j = 0; j < 8; ++j) a1[b].u[j] = hi ? 0u : pk2h(wr[2*j], wr[2*j+1]);
    }
    // fc2: D[j][t] = sum_d w2[j][d]*h[t][d];  A2_c[m=j][k=d-32c] = w2[j][d]
    HV a2[2];
    #pragma unroll
    for (int c = 0; c < 2; ++c) {
        const float* wr = w2 + lo16 * 64 + 32 * c + 16 * hi;
        #pragma unroll
        for (int j = 0; j < 4; ++j) {
            a2[c].u[j]     = pk2h(wr[2*j],      wr[2*j+1]);
            a2[c].u[j + 4] = pk2h(wr[8 + 2*j], wr[8 + 2*j + 1]);
        }
    }
    // biases in C/D layout: VGPR r, lane l -> row m = r + 8*(l>=16)
    float c1b[4][8];
    #pragma unroll
    for (int b = 0; b < 4; ++b)
        #pragma unroll
        for (int r = 0; r < 8; ++r)
            c1b[b][r] = b1[16*b + r + 8*hi];
    float c2b[8];
    #pragma unroll
    for (int r = 0; r < 8; ++r) c2b[r] = b2[r + 8*hi];

    float g[16], be[16];
    #pragma unroll
    for (int hh = 0; hh < 16; ++hh) { g[hh] = ln_g[hh]; be[hh] = ln_b[hh]; }

    const int nchunks = (n_tokens + 31) >> 5;
    const int nwaves  = (int)gridDim.x * 8;

    for (int chunk = (int)blockIdx.x * 8 + wv; chunk < nchunks; chunk += nwaves) {
        // ---- load: each lane owns one full token row (coalesced 2KB/wave) ----
        int tok = chunk * 32 + lane;
        int tl  = tok < n_tokens ? tok : (n_tokens - 1);
        float xr[16];
        const float4* xp = (const float4*)(x + (size_t)tl * 16);
        #pragma unroll
        for (int q = 0; q < 4; ++q) {
            float4 v = xp[q];
            xr[4*q+0] = v.x; xr[4*q+1] = v.y; xr[4*q+2] = v.z; xr[4*q+3] = v.w;
        }

        // ---- LayerNorm, fully in-lane ----
        float s = 0.f, ss = 0.f;
        #pragma unroll
        for (int hh = 0; hh < 16; ++hh) { s += xr[hh]; ss += xr[hh]*xr[hh]; }
        float mean = s * 0.0625f;
        float var  = ss * 0.0625f - mean * mean;
        float rstd = rsqrtf(var + 1e-5f);

        // yp: 16 f16 of this lane's normalized token, packed in pairs.
        // yq: tile-1 fc1 B operand (lanes 0-15 <- partner token; upper lanes
        //     hold K>=16 rows where A is zero, value irrelevant).
        uint32_t yp[8], yq[8];
        #pragma unroll
        for (int j = 0; j < 8; ++j) {
            float y0 = (xr[2*j]   - mean) * rstd * g[2*j]   + be[2*j];
            float y1 = (xr[2*j+1] - mean) * rstd * g[2*j+1] + be[2*j+1];
            yp[j] = pkrtz(y0, y1);
        }
        #pragma unroll
        for (int j = 0; j < 8; ++j) {
            uint32_t a = yp[j], b = yp[j];
            plswap(a, b);
            yq[j] = b;
        }

        // Residuals for both tiles in fc2 D layout (row m = r + 8*hi, col t = lo16):
        // one permlane swap per row yields both tiles' values.
        float res[2][8];
        #pragma unroll
        for (int r = 0; r < 8; ++r) {
            uint32_t a = f2u(xr[r]), b = f2u(xr[8 + r]);
            plswap(a, b);           // a -> tile0 residual, b -> tile1 residual
            res[0][r] = u2f(a);
            res[1][r] = u2f(b);
        }

        #pragma unroll
        for (int tile = 0; tile < 2; ++tile) {
            HV bv;
            #pragma unroll
            for (int j = 0; j < 8; ++j) bv.u[j] = tile ? yq[j] : yp[j];

            // fc1: 4 WMMAs -> h[d][t] (d = 16b + r + 8*hi, t = lo16);
            // epilogue = pack f16 + packed ReLU (1 cvt + 1 pk_max per pair)
            uint32_t p[4][4];
            #pragma unroll
            for (int b = 0; b < 4; ++b) {
                v8f acc;
                #pragma unroll
                for (int r = 0; r < 8; ++r) acc[r] = c1b[b][r];
                acc = __builtin_amdgcn_wmma_f32_16x16x32_f16(
                    false, a1[b].h, false, bv.h, (short)0, acc, false, false);
                #pragma unroll
                for (int j = 0; j < 4; ++j)
                    p[b][j] = relu_pk16(pkrtz(acc[2*j], acc[2*j+1]));
            }

            // fc2 B = h^T: C/D layout -> B layout is a half-wave merge; one
            // v_permlane16_swap per dword pair yields both B halves.
            HV b2v[2];
            #pragma unroll
            for (int c = 0; c < 2; ++c)
                #pragma unroll
                for (int j = 0; j < 4; ++j) {
                    uint32_t a = p[2*c][j], b = p[2*c + 1][j];
                    plswap(a, b);
                    b2v[c].u[j]     = a;
                    b2v[c].u[j + 4] = b;
                }

            // accumulator init (bias + residual) computed in place
            v8f d2;
            #pragma unroll
            for (int r = 0; r < 8; ++r) d2[r] = c2b[r] + res[tile][r];
            d2 = __builtin_amdgcn_wmma_f32_16x16x32_f16(
                false, a2[0].h, false, b2v[0].h, (short)0, d2, false, false);
            d2 = __builtin_amdgcn_wmma_f32_16x16x32_f16(
                false, a2[1].h, false, b2v[1].h, (short)0, d2, false, false);

            // ---- store (coalesced, two b128 per lane) ----
            int otok = chunk * 32 + tile * 16 + lo16;
            if (otok < n_tokens) {
                float4* op = (float4*)(out + (size_t)otok * 16 + 8 * hi);
                op[0] = make_float4(d2[0], d2[1], d2[2], d2[3]);
                op[1] = make_float4(d2[4], d2[5], d2[6], d2[7]);
            }
        }
    }
}

extern "C" void kernel_launch(void* const* d_in, const int* in_sizes, int n_in,
                              void* d_out, int out_size, void* d_ws, size_t ws_size,
                              hipStream_t stream) {
    const float* x    = (const float*)d_in[0];
    const float* ln_g = (const float*)d_in[1];
    const float* ln_b = (const float*)d_in[2];
    const float* w1   = (const float*)d_in[3];
    const float* b1   = (const float*)d_in[4];
    const float* w2   = (const float*)d_in[5];
    const float* b2   = (const float*)d_in[6];
    float* out = (float*)d_out;

    long long n_tokens = (long long)in_sizes[0] / 16;   // H = 16
    long long nchunks  = (n_tokens + 31) / 32;
    // ~4 chunks per wave to amortize the weight->register setup; cap grid.
    long long want_blocks = (nchunks + 8 * 4 - 1) / (8 * 4);
    int blocks = (int)(want_blocks < 1 ? 1 : (want_blocks > 1024 ? 1024 : want_blocks));

    ffn_ln_mlp_kernel<<<blocks, 256, 0, stream>>>(
        x, ln_g, ln_b, w1, b1, w2, b2, out, (int)n_tokens);
}
